// ModifiedGCN_21157008900177
// MI455X (gfx1250) — matrix operand order repaired and checked by reading, not compiled, not run on hardware
//
#include <hip/hip_runtime.h>
#include <hip/hip_bf16.h>
#include <math.h>

// ---------------- problem constants (match reference) ----------------
#define NNODES 100000
#define NEDGES 1600000
#define FDIM   64
#define NG     128
#define GF     32
#define L1DIM  256
#define L2DIM  128

typedef __attribute__((ext_vector_type(16))) _Float16 v16h;
typedef __attribute__((ext_vector_type(8)))  float    v8f;

#define LEAKY(v) ((v) > 0.0f ? (v) : 0.01f * (v))

// ---------------- degree / norm precompute ----------------
__global__ void deg_init_kernel(float* __restrict__ dinv, int n) {
    int i = blockIdx.x * blockDim.x + threadIdx.x;
    if (i < n) dinv[i] = 1.0f;   // self loop contributes 1
}

__global__ void deg_count_kernel(const int* __restrict__ dst, float* __restrict__ dinv, int e) {
    int i = blockIdx.x * blockDim.x + threadIdx.x;
    if (i < e) atomicAdd(&dinv[dst[i]], 1.0f);
}

__global__ void deg_finish_kernel(float* __restrict__ dinv, int n) {
    int i = blockIdx.x * blockDim.x + threadIdx.x;
    if (i < n) dinv[i] = rsqrtf(dinv[i]);   // deg >= 1 always (self loop)
}

__global__ void norm_kernel(const int* __restrict__ src, const int* __restrict__ dst,
                            const float* __restrict__ dinv, float* __restrict__ norm, int e) {
    int i = blockIdx.x * blockDim.x + threadIdx.x;
    if (i < e) norm[i] = dinv[src[i]] * dinv[dst[i]];
}

// ---------------- WMMA GEMM: H[n,64] = X[n,64] @ W[64,64] ----------------
// one wave per 16-row tile; K=64 -> 2x v_wmma_f32_16x16x32_f16 per 16-col tile
__global__ __launch_bounds__(256) void gemm64_wmma_kernel(
    const float* __restrict__ X, const float* __restrict__ W,
    float* __restrict__ H, int nrows)
{
    __shared__ _Float16 sWt[64 * 64];       // transposed: sWt[n*64 + k]
    const int t = threadIdx.x;
    for (int i = t; i < 4096; i += 256) {
        int k = i >> 6, n = i & 63;
        sWt[n * 64 + k] = (_Float16)W[i];
    }
    __syncthreads();

    const int wave = t >> 5;
    const int lane = t & 31;
    const int tile = blockIdx.x * 8 + wave;
    if (tile * 16 >= nrows) return;

    const int row  = tile * 16 + (lane & 15);
    const int koff = (lane >> 4) * 16;      // A-matrix K base for this lane half
    const int kb   = (lane >> 4) * 16;      // B-matrix K base for this lane half

    // A fragments (f32 -> f16), contiguous K order per lane half (matched by B):
    // lanes 0-15: K {0..15}; lanes 16-31: K {16..31}
    const float4* x4 = (const float4*)(X + (size_t)row * 64);
    float xa[16], xb[16];
    ((float4*)xa)[0] = x4[(koff >> 2) + 0];
    ((float4*)xa)[1] = x4[(koff >> 2) + 1];
    ((float4*)xa)[2] = x4[(koff >> 2) + 2];
    ((float4*)xa)[3] = x4[(koff >> 2) + 3];
    ((float4*)xb)[0] = x4[(koff >> 2) + 8];
    ((float4*)xb)[1] = x4[(koff >> 2) + 9];
    ((float4*)xb)[2] = x4[(koff >> 2) + 10];
    ((float4*)xb)[3] = x4[(koff >> 2) + 11];
    v16h a0, a1;
#pragma unroll
    for (int j = 0; j < 16; ++j) {
        a0[j] = (_Float16)xa[j];            // K 0..31
        a1[j] = (_Float16)xb[j];            // K 32..63
    }

    const int col = lane & 15;
#pragma unroll
    for (int ct = 0; ct < 4; ++ct) {
        const int ncol = ct * 16 + col;
        const _Float16* bp = &sWt[ncol * 64];
        v16h b0 = *(const v16h*)(bp + kb);        // K 0..31  (lane-half split)
        v16h b1 = *(const v16h*)(bp + 32 + kb);   // K 32..63
        v8f c = {};
        c = __builtin_amdgcn_wmma_f32_16x16x32_f16(false, a0, false, b0,
                                                   (short)0, c, false, false);
        c = __builtin_amdgcn_wmma_f32_16x16x32_f16(false, a1, false, b1,
                                                   (short)0, c, false, false);
        // C/D layout: VGPR v -> M = v (lanes 0-15) / v+8 (lanes 16-31), N = lane&15
        const int mbase = tile * 16 + ((lane >> 4) * 8);
#pragma unroll
        for (int v = 0; v < 8; ++v)
            H[(size_t)(mbase + v) * 64 + ncol] = c[v];
    }
}

// ---------------- aggregation ----------------
// agg[i,:] = h[i,:] * dinv[i]^2   (self-loop term; also initializes the buffer)
__global__ void agg_init_kernel(const float* __restrict__ H, const float* __restrict__ dinv,
                                float* __restrict__ AGG, int total /* = n*16 */) {
    int idx = blockIdx.x * blockDim.x + threadIdx.x;
    if (idx >= total) return;
    int i = idx >> 4;
    float s = dinv[i]; s = s * s;
    float4 h = ((const float4*)H)[idx];
    float4 o; o.x = h.x * s; o.y = h.y * s; o.z = h.z * s; o.w = h.w * s;
    ((float4*)AGG)[idx] = o;
}

// agg[dst,:] += norm[e] * h[src,:]   (16 threads per edge, float4 each)
__global__ void scatter_kernel(const int* __restrict__ src, const int* __restrict__ dst,
                               const float* __restrict__ norm, const float* __restrict__ H,
                               float* __restrict__ AGG, int total /* = e*16 */) {
    int idx = blockIdx.x * blockDim.x + threadIdx.x;
    if (idx >= total) return;
    int e = idx >> 4, q = idx & 15;
    int s = src[e], d = dst[e];
    float nm = norm[e];
    float4 h = *(const float4*)(H + (size_t)s * 64 + q * 4);
    float* out = AGG + (size_t)d * 64 + q * 4;
    atomicAdd(out + 0, nm * h.x);
    atomicAdd(out + 1, nm * h.y);
    atomicAdd(out + 2, nm * h.z);
    atomicAdd(out + 3, nm * h.w);
}

// x = leaky_relu(agg + b), in place
__global__ void finalize_kernel(float* __restrict__ AGG, const float* __restrict__ b,
                                int total /* = n*16 */) {
    int idx = blockIdx.x * blockDim.x + threadIdx.x;
    if (idx >= total) return;
    int f = (idx & 15) * 4;
    float4 v = ((const float4*)AGG)[idx];
    float4 bv = *(const float4*)(b + f);
    v.x = LEAKY(v.x + bv.x); v.y = LEAKY(v.y + bv.y);
    v.z = LEAKY(v.z + bv.z); v.w = LEAKY(v.w + bv.w);
    ((float4*)AGG)[idx] = v;
}

// ---------------- pooling ----------------
__global__ void pool_init_kernel(float* __restrict__ psum, float* __restrict__ pmax,
                                 float* __restrict__ pcnt) {
    int i = blockIdx.x * blockDim.x + threadIdx.x;
    if (i < NG * FDIM) { psum[i] = 0.0f; pmax[i] = -INFINITY; }
    if (i < NG) pcnt[i] = 0.0f;
}

__device__ inline void atomicMaxF(float* addr, float v) {
    if (v >= 0.0f) atomicMax((int*)addr, __float_as_int(v));
    else           atomicMin((unsigned int*)addr, __float_as_uint(v));
}

__global__ void pool_kernel(const float* __restrict__ X, const int* __restrict__ batch,
                            float* __restrict__ psum, float* __restrict__ pmax,
                            float* __restrict__ pcnt, int total /* = n*64 */) {
    int idx = blockIdx.x * blockDim.x + threadIdx.x;
    if (idx >= total) return;
    int i = idx >> 6, f = idx & 63;
    int g = batch[i];
    float v = X[idx];
    atomicAdd(&psum[g * FDIM + f], v);
    atomicMaxF(&pmax[g * FDIM + f], v);
    if (f == 0) atomicAdd(&pcnt[g], 1.0f);
}

// ---------------- MLP head ----------------
__global__ __launch_bounds__(256) void mlp1_kernel(
    const float* __restrict__ psum, const float* __restrict__ pmax,
    const float* __restrict__ pcnt, const float* __restrict__ gf,
    const float* __restrict__ W1, const float* __restrict__ b1,
    float* __restrict__ H1) {
    __shared__ float sP[FDIM * 2 + GF];     // 160
    int g = blockIdx.x, t = threadIdx.x;
    if (t < 160) {
        float v;
        if (t < 64)       v = psum[g * FDIM + t] / pcnt[g];
        else if (t < 128) v = pmax[g * FDIM + (t - 64)];
        else              v = gf[g * GF + (t - 128)];
        sP[t] = v;
    }
    __syncthreads();
    float acc = b1[t];
    for (int k = 0; k < 160; ++k) acc += sP[k] * W1[k * L1DIM + t];
    H1[g * L1DIM + t] = LEAKY(acc);
}

__global__ __launch_bounds__(128) void mlp2_kernel(
    const float* __restrict__ H1, const float* __restrict__ W2,
    const float* __restrict__ b2, float* __restrict__ H2) {
    __shared__ float sH[L1DIM];
    int g = blockIdx.x, t = threadIdx.x;
    sH[t]       = H1[g * L1DIM + t];
    sH[t + 128] = H1[g * L1DIM + t + 128];
    __syncthreads();
    float acc = b2[t];
    for (int k = 0; k < L1DIM; ++k) acc += sH[k] * W2[k * L2DIM + t];
    H2[g * L2DIM + t] = LEAKY(acc);
}

__global__ __launch_bounds__(128) void mlp3_kernel(
    const float* __restrict__ H2, const float* __restrict__ ow,
    const float* __restrict__ ob, float* __restrict__ out) {
    int g = threadIdx.x;     // 128 graphs
    float acc = ob[0];
    for (int k = 0; k < L2DIM; ++k) acc += H2[g * L2DIM + k] * ow[k];
    out[g] = acc;
}

// ---------------- host driver ----------------
extern "C" void kernel_launch(void* const* d_in, const int* in_sizes, int n_in,
                              void* d_out, int out_size, void* d_ws, size_t ws_size,
                              hipStream_t stream) {
    (void)in_sizes; (void)n_in; (void)out_size; (void)ws_size;
    const float* X0  = (const float*)d_in[0];
    const float* GFp = (const float*)d_in[1];
    const float* CW[3] = {(const float*)d_in[2], (const float*)d_in[4], (const float*)d_in[6]};
    const float* CB[3] = {(const float*)d_in[3], (const float*)d_in[5], (const float*)d_in[7]};
    const float* W1 = (const float*)d_in[8];
    const float* B1 = (const float*)d_in[9];
    const float* W2 = (const float*)d_in[10];
    const float* B2 = (const float*)d_in[11];
    const float* OW = (const float*)d_in[12];
    const float* OB = (const float*)d_in[13];
    const int*   EI = (const int*)d_in[14];
    const int*   BATCH = (const int*)d_in[15];
    const int* src = EI;
    const int* dst = EI + NEDGES;

    // workspace carve-up (floats)
    float* ws = (float*)d_ws;
    const size_t SZ_NF = (size_t)NNODES * FDIM;          // 6.4M floats
    float* H    = ws;
    float* AGGA = ws + SZ_NF;
    float* AGGB = ws + 2 * SZ_NF;
    float* DINV = ws + 3 * SZ_NF;                        // NNODES
    float* NORM = DINV + NNODES;                         // NEDGES
    float* PSUM = NORM + NEDGES;                         // NG*64
    float* PMAX = PSUM + NG * FDIM;
    float* PCNT = PMAX + NG * FDIM;                      // NG
    float* H1   = PCNT + NG;                             // NG*256
    float* H2   = H1 + NG * L1DIM;                       // NG*128

    const int TB = 256;
    // 1) degrees + per-edge norms (reused by all 3 layers)
    deg_init_kernel<<<(NNODES + TB - 1) / TB, TB, 0, stream>>>(DINV, NNODES);
    deg_count_kernel<<<(NEDGES + TB - 1) / TB, TB, 0, stream>>>(dst, DINV, NEDGES);
    deg_finish_kernel<<<(NNODES + TB - 1) / TB, TB, 0, stream>>>(DINV, NNODES);
    norm_kernel<<<(NEDGES + TB - 1) / TB, TB, 0, stream>>>(src, dst, DINV, NORM, NEDGES);

    // 2) three GCN conv layers (ping-pong AGGA/AGGB as layer outputs)
    const int n16 = NNODES * 16, e16 = NEDGES * 16;
    const int gemm_blocks = ((NNODES / 16) + 7) / 8;     // one wave per 16-row tile
    const float* xin = X0;
    float* xout[3] = {AGGA, AGGB, AGGA};
    for (int l = 0; l < 3; ++l) {
        gemm64_wmma_kernel<<<gemm_blocks, TB, 0, stream>>>(xin, CW[l], H, NNODES);
        agg_init_kernel<<<(n16 + TB - 1) / TB, TB, 0, stream>>>(H, DINV, xout[l], n16);
        scatter_kernel<<<(e16 + TB - 1) / TB, TB, 0, stream>>>(src, dst, NORM, H, xout[l], e16);
        finalize_kernel<<<(n16 + TB - 1) / TB, TB, 0, stream>>>(xout[l], CB[l], n16);
        xin = xout[l];
    }

    // 3) mean/max pooling per graph
    pool_init_kernel<<<(NG * FDIM + TB - 1) / TB, TB, 0, stream>>>(PSUM, PMAX, PCNT);
    pool_kernel<<<(NNODES * FDIM + TB - 1) / TB, TB, 0, stream>>>(xout[2], BATCH, PSUM, PMAX, PCNT,
                                                                  NNODES * FDIM);

    // 4) MLP head -> d_out [128]
    mlp1_kernel<<<NG, L1DIM, 0, stream>>>(PSUM, PMAX, PCNT, GFp, W1, B1, H1);
    mlp2_kernel<<<NG, L2DIM, 0, stream>>>(H1, W2, B2, H2);
    mlp3_kernel<<<1, NG, 0, stream>>>(H2, OW, OB, (float*)d_out);
}
